// Mamba2Blockv2_68161130987997
// MI455X (gfx1250) — matrix operand order repaired and checked by reading, not compiled, not run on hardware
//
#include <hip/hip_runtime.h>

#define DM     1024
#define DI     2048
#define DS     128
#define HD     64
#define NH     32
#define DCONV  4
#define CHUNKL 256
#define CONVD  2304          // DI + 2*DS
#define DPROJ  4384          // 2*DI + 2*DS + NH
#define SEQL   4096
#define BATCH  2
#define BL     (BATCH*SEQL)  // 8192
#define NCHUNK (SEQL/CHUNKL) // 16
#define EPSF   1e-5f

typedef __attribute__((ext_vector_type(16))) __bf16 v16bf;
typedef __attribute__((ext_vector_type(8)))  float v8f;
typedef int v4i __attribute__((vector_size(16)));

union FragA { v16bf v; unsigned int u2[8]; unsigned short u[16]; };

#define AS1 __attribute__((address_space(1)))
#define AS3 __attribute__((address_space(3)))

#if __has_builtin(__builtin_amdgcn_global_load_async_to_lds_b128)
#define HAVE_ASYNC 1
#else
#define HAVE_ASYNC 0
#endif

// 16B memory->LDS copy: async (ASYNCcnt) on CDNA5, sync fallback otherwise.
__device__ __forceinline__ void cp16(void* lds, const void* g) {
#if HAVE_ASYNC
  __builtin_amdgcn_global_load_async_to_lds_b128(
      (AS1 v4i*)(unsigned long long)g,
      (AS3 v4i*)(unsigned int)(unsigned long long)lds, 0, 0);
#else
  *(uint4*)lds = *(const uint4*)g;
#endif
}

template <int N>
__device__ __forceinline__ void async_waitN() {
#if HAVE_ASYNC
#if __has_builtin(__builtin_amdgcn_s_wait_asynccnt)
  __builtin_amdgcn_s_wait_asynccnt((unsigned short)N);
#else
  asm volatile("s_wait_asynccnt %0" :: "i"(N) : "memory");
#endif
#endif
}

__device__ __forceinline__ unsigned short f2bf(float f) {
  unsigned int u = __float_as_uint(f);
  u += 0x7FFFu + ((u >> 16) & 1u);
  return (unsigned short)(u >> 16);
}
__device__ __forceinline__ float bf2f(unsigned short h) {
  return __uint_as_float(((unsigned int)h) << 16);
}
__device__ __forceinline__ v8f wmma_bf16(v16bf a, v16bf b, v8f c) {
  return __builtin_amdgcn_wmma_f32_16x16x32_bf16(false, a, false, b, (short)0, c, false, false);
}
__device__ __forceinline__ v8f vzero8() { v8f z = {0.f,0.f,0.f,0.f,0.f,0.f,0.f,0.f}; return z; }

// ---------------- RMSNorm (input) -> bf16 ----------------
__global__ __launch_bounds__(256) void k_rmsnorm_in(const float* __restrict__ x,
                                                    const float* __restrict__ w,
                                                    unsigned short* __restrict__ hb) {
  int row = blockIdx.x;
  const float* xr = x + (size_t)row * DM;
  __shared__ float red[8];
  float s = 0.f;
  for (int j = threadIdx.x; j < DM; j += 256) { float v = xr[j]; s += v * v; }
  for (int off = 16; off; off >>= 1) s += __shfl_down(s, off, 32);
  int wave = threadIdx.x >> 5, lane = threadIdx.x & 31;
  if (lane == 0) red[wave] = s;
  __syncthreads();
  if (wave == 0) {
    float t = (lane < 8) ? red[lane] : 0.f;
    for (int off = 4; off; off >>= 1) t += __shfl_down(t, off, 32);
    if (lane == 0) red[0] = t;
  }
  __syncthreads();
  float rs = rsqrtf(red[0] / (float)DM + EPSF);
  for (int j = threadIdx.x; j < DM; j += 256)
    hb[(size_t)row * DM + j] = f2bf(xr[j] * rs * w[j]);
}

// ---------------- f32 -> bf16 convert ----------------
__global__ __launch_bounds__(256) void k_f2bf(const float* __restrict__ src,
                                              unsigned short* __restrict__ dst, int n) {
  int i = blockIdx.x * 256 + threadIdx.x;
  if (i < n) dst[i] = f2bf(src[i]);
}

// ---------------- generic bf16 WMMA GEMM: C[M,N] = A[M,K] * Bw[N,K]^T (+res) ----
// Double-buffered LDS tiles filled with GLOBAL_LOAD_ASYNC_TO_LDS_B128; the next
// tile's fill overlaps the current tile's 8 WMMAs (ASYNCcnt pipelining).
__global__ __launch_bounds__(256) void k_gemm_bf16(const unsigned short* __restrict__ A,
                                                   const unsigned short* __restrict__ Bw,
                                                   float* __restrict__ Cout,
                                                   int M, int N, int K, int ldc,
                                                   const float* __restrict__ res) {
  __shared__ unsigned short sA[2][128][32];
  __shared__ unsigned short sB[2][128][32];
  int bm = blockIdx.y * 128, bn = blockIdx.x * 128;
  int tid = threadIdx.x, lane = tid & 31, wave = tid >> 5;
  int lm = lane & 15, hi = (lane >= 16);
  int wm = (wave & 3) * 32;   // 4 waves over 128 rows
  int wn = (wave >> 2) * 64;  // 2 waves over 128 cols
  v8f acc[8];
  #pragma unroll
  for (int i = 0; i < 8; i++) acc[i] = vzero8();
  int lrow = tid >> 1, lcol = (tid & 1) * 16;

  auto issue = [&](int buf, int k0) {
    const unsigned short* srcA = A + (size_t)(bm + lrow) * K + k0 + lcol;
    cp16(&sA[buf][lrow][lcol],     srcA);
    cp16(&sA[buf][lrow][lcol + 8], srcA + 8);
    int n = bn + lrow;
    if (n < N) {
      const unsigned short* srcB = Bw + (size_t)n * K + k0 + lcol;
      cp16(&sB[buf][lrow][lcol],     srcB);
      cp16(&sB[buf][lrow][lcol + 8], srcB + 8);
    } else {
      uint4 z = {0u,0u,0u,0u};
      *(uint4*)&sB[buf][lrow][lcol] = z;
      *(uint4*)&sB[buf][lrow][lcol + 8] = z;
    }
  };

  issue(0, 0);
  int cur = 0;
  for (int k0 = 0; k0 < K; k0 += 32) {
    if (k0 + 32 < K) {
      issue(cur ^ 1, k0 + 32);   // prefetch next tile into other buffer
      async_waitN<4>();          // current tile done; next tile may stay in flight
    } else {
      async_waitN<0>();
    }
    __syncthreads();
    FragA a[2], b;
    #pragma unroll
    for (int mt = 0; mt < 2; mt++) {
      int m = wm + mt * 16 + lm;
      int kb = hi ? 16 : 0;
      #pragma unroll
      for (int j = 0; j < 4; j++) a[mt].u2[j]     = *(const unsigned int*)&sA[cur][m][kb + 2*j];
      #pragma unroll
      for (int j = 0; j < 4; j++) a[mt].u2[4 + j] = *(const unsigned int*)&sA[cur][m][kb + 8 + 2*j];
    }
    #pragma unroll
    for (int nt = 0; nt < 4; nt++) {
      int n = wn + nt * 16 + lm;
      int kb = hi ? 16 : 0;
      #pragma unroll
      for (int j = 0; j < 8; j++) b.u2[j] = *(const unsigned int*)&sB[cur][n][kb + 2*j];
      #pragma unroll
      for (int mt = 0; mt < 2; mt++)
        acc[mt * 4 + nt] = wmma_bf16(a[mt].v, b.v, acc[mt * 4 + nt]);
    }
    __syncthreads();
    cur ^= 1;
  }
  int rbase = hi ? 8 : 0;
  #pragma unroll
  for (int mt = 0; mt < 2; mt++)
    #pragma unroll
    for (int nt = 0; nt < 4; nt++) {
      int n = bn + wn + nt * 16 + lm;
      if (n < N) {
        #pragma unroll
        for (int r = 0; r < 8; r++) {
          int m = bm + wm + mt * 16 + rbase + r;
          size_t off = (size_t)m * ldc + n;
          float v = acc[mt * 4 + nt][r];
          if (res) v += res[off];
          Cout[off] = v;
        }
      }
    }
}

// ---------------- depthwise conv(4) + SiLU + split ----------------
__global__ __launch_bounds__(256) void k_conv(const float* __restrict__ zx,
                                              const float* __restrict__ cw,
                                              const float* __restrict__ cb,
                                              float* __restrict__ xsf,
                                              unsigned short* __restrict__ Bbf,
                                              unsigned short* __restrict__ Cbf) {
  int blk = blockIdx.x;               // 1024 blocks, 8 rows each
  int b  = (blk * 8) / SEQL;
  int l0 = (blk * 8) % SEQL;
  for (int ch = threadIdx.x; ch < CONVD; ch += 256) {
    float w0 = cw[ch*4+0], w1 = cw[ch*4+1], w2 = cw[ch*4+2], w3 = cw[ch*4+3];
    float bias = cb[ch];
    size_t colbase = ((size_t)b * SEQL) * DPROJ + (size_t)DI + ch;
    for (int i = 0; i < 8; i++) {
      int l = l0 + i;
      float s = bias;
      s += (l >= 3 ? zx[colbase + (size_t)(l-3) * DPROJ] : 0.f) * w0;
      s += (l >= 2 ? zx[colbase + (size_t)(l-2) * DPROJ] : 0.f) * w1;
      s += (l >= 1 ? zx[colbase + (size_t)(l-1) * DPROJ] : 0.f) * w2;
      s += zx[colbase + (size_t)l * DPROJ] * w3;
      float v = s / (1.f + __expf(-s));
      size_t row = (size_t)b * SEQL + l;
      if (ch < DI)            xsf[row * DI + ch] = v;
      else if (ch < DI + DS)  Bbf[row * DS + (ch - DI)] = f2bf(v);
      else                    Cbf[row * DS + (ch - DI - DS)] = f2bf(v);
    }
  }
}

// ---------------- dt softplus, dt*A ----------------
__global__ __launch_bounds__(256) void k_dt(const float* __restrict__ zx,
                                            const float* __restrict__ dt_bias,
                                            const float* __restrict__ A_log,
                                            float* __restrict__ dtv, float* __restrict__ dtA) {
  int i = blockIdx.x * 256 + threadIdx.x;
  if (i >= BL * NH) return;
  int h = i & (NH - 1); int row = i >> 5;
  float x = zx[(size_t)row * DPROJ + (DI + CONVD) + h] + dt_bias[h];
  float sp = (x > 20.f) ? x : log1pf(__expf(x));
  float A = -__expf(A_log[h]);
  dtv[i] = sp;
  dtA[i] = sp * A;
}

// ---------------- per-chunk inclusive cumsum of dtA ----------------
__global__ __launch_bounds__(256) void k_cumsum(const float* __restrict__ dtA,
                                                float* __restrict__ Acs) {
  int bid = blockIdx.x;               // (b*NCHUNK + c)*NH + h
  int h = bid % NH; int c = (bid / NH) % NCHUNK; int b = bid / (NH * NCHUNK);
  int tid = threadIdx.x;
  __shared__ float buf[256];
  size_t row = (size_t)b * SEQL + (size_t)c * CHUNKL + tid;
  buf[tid] = dtA[row * NH + h];
  __syncthreads();
  for (int off = 1; off < 256; off <<= 1) {
    float v = (tid >= off) ? buf[tid - off] : 0.f;
    __syncthreads();
    buf[tid] += v;
    __syncthreads();
  }
  Acs[(size_t)bid * 256 + tid] = buf[tid];
}

// ---------------- xd = xs * dt -> bf16 ----------------
__global__ __launch_bounds__(256) void k_xd(const float* __restrict__ xsf,
                                            const float* __restrict__ dtv,
                                            unsigned short* __restrict__ xdbf) {
  size_t i = (size_t)blockIdx.x * 256 + threadIdx.x;
  if (i >= (size_t)BL * DI) return;
  int col = (int)(i & (DI - 1)); int h = col >> 6; size_t row = i >> 11;
  xdbf[i] = f2bf(xsf[i] * dtv[row * NH + h]);
}

// ---------------- SSD intra-chunk: y_diag + states, per (b,c,h) ----------------
__global__ __launch_bounds__(256) void k_ssd_intra(const unsigned short* __restrict__ Bbf,
                                                   const unsigned short* __restrict__ Cbf,
                                                   const unsigned short* __restrict__ xdbf,
                                                   const float* __restrict__ Acs,
                                                   float* __restrict__ ydiag,
                                                   float* __restrict__ Sbuf) {
  __shared__ __align__(16) unsigned char smem[54272];
  float* sAcs = (float*)smem;                                                 // 1 KB
  unsigned short (*sB)[32]   = (unsigned short(*)[32])(smem + 1024);          // [256][32] 16 KB
  unsigned short (*sC)[32]   = (unsigned short(*)[32])(smem + 1024 + 16384);  // [64][32]  4 KB
  unsigned short (*sX)[64]   = (unsigned short(*)[64])(smem + 1024 + 16384);  // alias sC
  unsigned short (*sG)[256]  = (unsigned short(*)[256])(smem + 1024 + 16384 + 4096); // 32 KB
  unsigned short (*sBt)[128] = (unsigned short(*)[128])(smem + 1024);         // alias sB
  unsigned short (*sXs)[64]  = (unsigned short(*)[64])(smem + 1024 + 8192);   // alias sB tail

  int bid = blockIdx.x;
  int h = bid % NH; int c = (bid / NH) % NCHUNK; int b = bid / (NH * NCHUNK);
  size_t r0 = (size_t)b * SEQL + (size_t)c * CHUNKL;
  int tid = threadIdx.x, lane = tid & 31, wave = tid >> 5;
  int lm = lane & 15, hi = (lane >= 16);

  sAcs[tid] = Acs[(size_t)bid * 256 + tid];
  __syncthreads();
  float Alast = sAcs[255];

  for (int lt = 0; lt < 4; ++lt) {
    // G = Cc * Bc^T for 64 rows
    int gwm = (wave & 3) * 16;
    int gwn = (wave >> 2) * 128;
    v8f g[8];
    #pragma unroll
    for (int i = 0; i < 8; i++) g[i] = vzero8();
    for (int kk = 0; kk < 4; ++kk) {
      {
        const unsigned short* src = Bbf + (r0 + tid) * DS + kk * 32;
        cp16(&sB[tid][0],  src);
        cp16(&sB[tid][8],  src + 8);
        cp16(&sB[tid][16], src + 16);
        cp16(&sB[tid][24], src + 24);
      }
      if (tid < 128) {
        int rr = tid >> 1, half = (tid & 1) * 16;
        const unsigned short* src = Cbf + (r0 + lt * 64 + rr) * DS + kk * 32 + half;
        cp16(&sC[rr][half],     src);
        cp16(&sC[rr][half + 8], src + 8);
      }
      async_waitN<0>();
      __syncthreads();
      FragA a, bf;
      {
        int m = gwm + lm; int kb = hi ? 16 : 0;
        #pragma unroll
        for (int j = 0; j < 4; j++) a.u2[j]     = *(const unsigned int*)&sC[m][kb + 2*j];
        #pragma unroll
        for (int j = 0; j < 4; j++) a.u2[4 + j] = *(const unsigned int*)&sC[m][kb + 8 + 2*j];
      }
      #pragma unroll
      for (int nt = 0; nt < 8; ++nt) {
        int sI = gwn + nt * 16 + lm;
        int kb = hi ? 16 : 0;
        #pragma unroll
        for (int j = 0; j < 8; j++) bf.u2[j] = *(const unsigned int*)&sB[sI][kb + 2*j];
        g[nt] = wmma_bf16(a.v, bf.v, g[nt]);
      }
      __syncthreads();
    }
    // causal decay mask -> bf16 in sG
    {
      int rbase = hi ? 8 : 0;
      #pragma unroll
      for (int nt = 0; nt < 8; ++nt) {
        int sI = gwn + nt * 16 + lm;
        float as = sAcs[sI];
        #pragma unroll
        for (int r = 0; r < 8; r++) {
          int lrow = gwm + rbase + r;
          int l = lt * 64 + lrow;
          float v = g[nt][r];
          v = (sI <= l) ? v * __expf(sAcs[l] - as) : 0.f;
          sG[lrow][sI] = f2bf(v);
        }
      }
    }
    __syncthreads();
    // Y_diag rows = sG(64x256) * xd(256x64)
    int ywm = (wave >> 1) * 16;
    int ywn = (wave & 1) * 32;
    v8f y[2]; y[0] = vzero8(); y[1] = vzero8();
    for (int kk = 0; kk < 8; ++kk) {
      {
        int rr = tid >> 3, cc = (tid & 7) * 8;
        const unsigned short* src = xdbf + (r0 + kk * 32 + rr) * DI + h * 64 + cc;
        cp16(&sX[rr][cc], src);
      }
      async_waitN<0>();
      __syncthreads();
      FragA a, bf;
      {
        int m = ywm + lm; int kb = kk * 32 + (hi ? 16 : 0);
        #pragma unroll
        for (int j = 0; j < 4; j++) a.u2[j]     = *(const unsigned int*)&sG[m][kb + 2*j];
        #pragma unroll
        for (int j = 0; j < 4; j++) a.u2[4 + j] = *(const unsigned int*)&sG[m][kb + 8 + 2*j];
      }
      #pragma unroll
      for (int nt = 0; nt < 2; ++nt) {
        int p = ywn + nt * 16 + lm;
        int kb = hi ? 16 : 0;
        #pragma unroll
        for (int e = 0; e < 16; e++) bf.u[e] = sX[kb + e][p];
        y[nt] = wmma_bf16(a.v, bf.v, y[nt]);
      }
      __syncthreads();
    }
    {
      int rbase = hi ? 8 : 0;
      #pragma unroll
      for (int nt = 0; nt < 2; ++nt) {
        int p = ywn + nt * 16 + lm;
        #pragma unroll
        for (int r = 0; r < 8; r++) {
          int l = lt * 64 + ywm + rbase + r;
          ydiag[(r0 + l) * DI + h * 64 + p] = y[nt][r];
        }
      }
    }
    __syncthreads();
  }

  // states S[n,p] = sum_l Bc[l,n]*exp(Alast-Acs[l])*xd[l,p]
  v8f sa[4];
  #pragma unroll
  for (int i = 0; i < 4; i++) sa[i] = vzero8();
  int nblk = wave * 16;
  for (int kk = 0; kk < 8; ++kk) {
    {
      int rr = tid >> 3, cc = (tid & 7) * 16;
      const unsigned short* src = Bbf + (r0 + kk * 32 + rr) * DS + cc;
      cp16(&sBt[rr][cc],     src);
      cp16(&sBt[rr][cc + 8], src + 8);
    }
    {
      int rr = tid >> 3, cc = (tid & 7) * 8;
      const unsigned short* src = xdbf + (r0 + kk * 32 + rr) * DI + h * 64 + cc;
      cp16(&sXs[rr][cc], src);
    }
    async_waitN<0>();
    __syncthreads();
    FragA a, bf;
    {
      int n = nblk + lm;
      #pragma unroll
      for (int e = 0; e < 16; e++) {
        int ll = e + (hi ? 16 : 0);
        float dec = __expf(Alast - sAcs[kk * 32 + ll]);
        a.u[e] = f2bf(bf2f(sBt[ll][n]) * dec);
      }
    }
    #pragma unroll
    for (int nt = 0; nt < 4; ++nt) {
      int p = nt * 16 + lm;
      int kb = hi ? 16 : 0;
      #pragma unroll
      for (int e = 0; e < 16; e++) bf.u[e] = sXs[kb + e][p];
      sa[nt] = wmma_bf16(a.v, bf.v, sa[nt]);
    }
    __syncthreads();
  }
  {
    int rbase = hi ? 8 : 0;
    size_t sbase = (size_t)bid * (DS * HD);
    #pragma unroll
    for (int nt = 0; nt < 4; ++nt) {
      int p = nt * 16 + lm;
      #pragma unroll
      for (int r = 0; r < 8; r++) {
        int n = nblk + rbase + r;
        Sbuf[sbase + (size_t)n * HD + p] = sa[nt][r];
      }
    }
  }
}

// ---------------- inter-chunk scan ----------------
__global__ __launch_bounds__(256) void k_scan(const float* __restrict__ Sbuf,
                                              const float* __restrict__ Acs,
                                              unsigned short* __restrict__ prevbf) {
  int bid = blockIdx.x;   // b*NH + h
  int h = bid % NH, b = bid / NH;
  __shared__ float dec[NCHUNK];
  if (threadIdx.x < NCHUNK) {
    int c = threadIdx.x;
    dec[c] = __expf(Acs[(((size_t)(b * NCHUNK + c)) * NH + h) * 256 + 255]);
  }
  __syncthreads();
  for (int e = threadIdx.x; e < DS * HD; e += 256) {
    float carry = 0.f;
    for (int c = 0; c < NCHUNK; ++c) {
      size_t idx = ((size_t)((b * NCHUNK + c) * NH + h)) * (DS * HD) + e;
      prevbf[idx] = f2bf(carry);
      carry = carry * dec[c] + Sbuf[idx];
    }
  }
}

// ---------------- y_off + decay + D*x combine (in place over ydiag) ----------------
__global__ __launch_bounds__(256) void k_ssd_off(const unsigned short* __restrict__ Cbf,
                                                 const unsigned short* __restrict__ prevbf,
                                                 const float* __restrict__ Acs,
                                                 const float* __restrict__ xsf,
                                                 const float* __restrict__ Dp,
                                                 float* __restrict__ y) {
  __shared__ float sAcs[256];
  __shared__ unsigned short sC[256][32];
  __shared__ unsigned short sP[32][64];
  int bid = blockIdx.x;
  int h = bid % NH; int c = (bid / NH) % NCHUNK; int b = bid / (NH * NCHUNK);
  size_t r0 = (size_t)b * SEQL + (size_t)c * CHUNKL;
  int tid = threadIdx.x, lane = tid & 31, wave = tid >> 5;
  int lm = lane & 15, hi = (lane >= 16);
  sAcs[tid] = Acs[(size_t)bid * 256 + tid];
  v8f o[2][4];
  #pragma unroll
  for (int i = 0; i < 2; i++)
    #pragma unroll
    for (int j = 0; j < 4; j++) o[i][j] = vzero8();
  size_t pbase = (size_t)bid * (DS * HD);
  for (int kk = 0; kk < 4; ++kk) {
    {
      const unsigned short* src = Cbf + (r0 + tid) * DS + kk * 32;
      cp16(&sC[tid][0],  src);
      cp16(&sC[tid][8],  src + 8);
      cp16(&sC[tid][16], src + 16);
      cp16(&sC[tid][24], src + 24);
    }
    {
      int rr = tid >> 3, cc = (tid & 7) * 8;
      cp16(&sP[rr][cc], &prevbf[pbase + (size_t)(kk * 32 + rr) * HD + cc]);
    }
    async_waitN<0>();
    __syncthreads();
    FragA a, bf;
    #pragma unroll
    for (int mt = 0; mt < 2; ++mt) {
      int l = (wave * 2 + mt) * 16 + lm;
      int kb = hi ? 16 : 0;
      #pragma unroll
      for (int j = 0; j < 4; j++) a.u2[j]     = *(const unsigned int*)&sC[l][kb + 2*j];
      #pragma unroll
      for (int j = 0; j < 4; j++) a.u2[4 + j] = *(const unsigned int*)&sC[l][kb + 8 + 2*j];
      #pragma unroll
      for (int nt = 0; nt < 4; ++nt) {
        int p = nt * 16 + lm;
        int kb2 = hi ? 16 : 0;
        #pragma unroll
        for (int e = 0; e < 16; e++) bf.u[e] = sP[kb2 + e][p];
        o[mt][nt] = wmma_bf16(a.v, bf.v, o[mt][nt]);
      }
    }
    __syncthreads();
  }
  int rbase = hi ? 8 : 0;
  float dph = Dp[h];
  #pragma unroll
  for (int mt = 0; mt < 2; ++mt)
    #pragma unroll
    for (int nt = 0; nt < 4; ++nt) {
      int p = nt * 16 + lm;
      #pragma unroll
      for (int r = 0; r < 8; r++) {
        int l = (wave * 2 + mt) * 16 + rbase + r;
        float sd = __expf(sAcs[l]);
        size_t off = (r0 + l) * DI + h * 64 + p;
        y[off] = y[off] + o[mt][nt][r] * sd + dph * xsf[off];
      }
    }
}

// ---------------- gated RMSNorm -> bf16 ----------------
__global__ __launch_bounds__(256) void k_gatenorm(const float* __restrict__ y,
                                                  const float* __restrict__ zx,
                                                  const float* __restrict__ nw,
                                                  unsigned short* __restrict__ ybf) {
  int row = blockIdx.x;
  const float* yr = y + (size_t)row * DI;
  const float* zr = zx + (size_t)row * DPROJ;
  __shared__ float red[8];
  float s = 0.f;
  for (int j = threadIdx.x; j < DI; j += 256) {
    float z = zr[j];
    float u = yr[j] * (z / (1.f + __expf(-z)));
    s += u * u;
  }
  for (int off = 16; off; off >>= 1) s += __shfl_down(s, off, 32);
  int wave = threadIdx.x >> 5, lane = threadIdx.x & 31;
  if (lane == 0) red[wave] = s;
  __syncthreads();
  if (wave == 0) {
    float t = (lane < 8) ? red[lane] : 0.f;
    for (int off = 4; off; off >>= 1) t += __shfl_down(t, off, 32);
    if (lane == 0) red[0] = t;
  }
  __syncthreads();
  float rs = rsqrtf(red[0] / (float)DI + EPSF);
  for (int j = threadIdx.x; j < DI; j += 256) {
    float z = zr[j];
    float u = yr[j] * (z / (1.f + __expf(-z)));
    ybf[(size_t)row * DI + j] = f2bf(u * rs * nw[j]);
  }
}

extern "C" void kernel_launch(void* const* d_in, const int* in_sizes, int n_in,
                              void* d_out, int out_size, void* d_ws, size_t ws_size,
                              hipStream_t stream) {
  const float* x        = (const float*)d_in[0];
  const float* rms_w    = (const float*)d_in[1];
  const float* in_pw    = (const float*)d_in[2];
  const float* conv_w   = (const float*)d_in[3];
  const float* conv_b   = (const float*)d_in[4];
  const float* dt_bias  = (const float*)d_in[5];
  const float* A_log    = (const float*)d_in[6];
  const float* Dp       = (const float*)d_in[7];
  const float* norm_w   = (const float*)d_in[8];
  const float* out_pw   = (const float*)d_in[9];
  float* out = (float*)d_out;

  char* ws = (char*)d_ws;
  size_t off = 0;
  auto carve = [&](size_t bytes) -> char* {
    off = (off + 255) & ~(size_t)255;
    char* p = ws + off;
    off += bytes;
    return p;
  };
  unsigned short* Hbf    = (unsigned short*)carve((size_t)BL * DM * 2);
  unsigned short* Wi     = (unsigned short*)carve((size_t)DPROJ * DM * 2);
  unsigned short* Wo     = (unsigned short*)carve((size_t)DM * DI * 2);
  float*          zx     = (float*)carve((size_t)BL * DPROJ * 4);
  float*          xsf    = (float*)carve((size_t)BL * DI * 4);
  unsigned short* Bbf    = (unsigned short*)carve((size_t)BL * DS * 2);
  unsigned short* Cbf    = (unsigned short*)carve((size_t)BL * DS * 2);
  float*          dtv    = (float*)carve((size_t)BL * NH * 4);
  float*          dtA    = (float*)carve((size_t)BL * NH * 4);
  float*          Acs    = (float*)carve((size_t)BATCH * NCHUNK * NH * 256 * 4);
  unsigned short* xdbf   = (unsigned short*)carve((size_t)BL * DI * 2);
  float*          ydiag  = (float*)carve((size_t)BL * DI * 4);
  float*          Sbuf   = (float*)carve((size_t)BATCH * NCHUNK * NH * DS * HD * 4);
  unsigned short* prevbf = (unsigned short*)carve((size_t)BATCH * NCHUNK * NH * DS * HD * 2);
  unsigned short* ybf    = (unsigned short*)carve((size_t)BL * DI * 2);
  (void)ws_size; (void)in_sizes; (void)n_in; (void)out_size;

  // 1. RMSNorm input -> bf16
  k_rmsnorm_in<<<BL, 256, 0, stream>>>(x, rms_w, Hbf);
  // 2. weight converts
  {
    int n1 = DPROJ * DM;
    k_f2bf<<<(n1 + 255) / 256, 256, 0, stream>>>(in_pw, Wi, n1);
    int n2 = DM * DI;
    k_f2bf<<<(n2 + 255) / 256, 256, 0, stream>>>(out_pw, Wo, n2);
  }
  // 3. in_proj GEMM: zx[8192,4384] = Hbf * Wi^T
  {
    dim3 grid((DPROJ + 127) / 128, BL / 128);
    k_gemm_bf16<<<grid, 256, 0, stream>>>(Hbf, Wi, zx, BL, DPROJ, DM, DPROJ, nullptr);
  }
  // 4. conv + SiLU + split
  k_conv<<<BL / 8, 256, 0, stream>>>(zx, conv_w, conv_b, xsf, Bbf, Cbf);
  // 5. dt softplus
  k_dt<<<(BL * NH + 255) / 256, 256, 0, stream>>>(zx, dt_bias, A_log, dtv, dtA);
  // 6. per-chunk cumsum
  k_cumsum<<<BATCH * NCHUNK * NH, 256, 0, stream>>>(dtA, Acs);
  // 7. xd = xs*dt
  k_xd<<<(int)(((size_t)BL * DI + 255) / 256), 256, 0, stream>>>(xsf, dtv, xdbf);
  // 8. intra-chunk SSD (y_diag + states)
  k_ssd_intra<<<BATCH * NCHUNK * NH, 256, 0, stream>>>(Bbf, Cbf, xdbf, Acs, ydiag, Sbuf);
  // 9. inter-chunk scan
  k_scan<<<BATCH * NH, 256, 0, stream>>>(Sbuf, Acs, prevbf);
  // 10. y_off + D*x combine (in place)
  k_ssd_off<<<BATCH * NCHUNK * NH, 256, 0, stream>>>(Cbf, prevbf, Acs, xsf, Dp, ydiag);
  // 11. gated RMSNorm -> bf16
  k_gatenorm<<<BL, 256, 0, stream>>>(ydiag, zx, norm_w, ybf);
  // 12. out_proj GEMM + residual
  {
    dim3 grid(DM / 128, BL / 128);
    k_gemm_bf16<<<grid, 256, 0, stream>>>(ybf, Wo, out, BL, DM, DI, DM, x);
  }
}
